// CKConv_22333829939292
// MI455X (gfx1250) — matrix operand order repaired and checked by reading, not compiled, not run on hardware
//
#include <hip/hip_runtime.h>
#include <math.h>

namespace {
constexpr int   H_      = 64;
constexpr int   E_      = 30000;
constexpr int   NU_     = 10000;
constexpr int   KH_     = 50;
constexpr float OMEGA_  = 30.0f;
constexpr int   KDIM_   = (KH_ + 1) * H_;   // 3264: j=0..49 from w3, j=50 = bias channel (b3)
constexpr int   NCHUNK_ = KDIM_ / 32;       // 102 K-chunks of 32 for wmma f16 16x16x32
constexpr int   EPAD_   = 30016;            // E padded to multiple of 32
constexpr int   NT32_   = (E_ + 31) / 32;   // 938 edge-groups of 32
}

typedef __attribute__((ext_vector_type(16))) _Float16 v16h;
typedef __attribute__((ext_vector_type(8)))  _Float16 v8h;
typedef __attribute__((ext_vector_type(8)))  float    v8f;

__global__ void zero_out(float* p, int n) {
  int i = blockIdx.x * 256 + threadIdx.x;
  if (i < n) p[i] = 0.f;
}

// Stage 1: per-edge SIREN kernel-net up to h2 (KH values), stored as f16 rows of 64
// (col 50 = 1.0 bias channel, cols 51..63 = 0).
__global__ __launch_bounds__(256) void knet_h2(
    const float* __restrict__ node_t, const float* __restrict__ edge_t,
    const int* __restrict__ nidx,
    const float* __restrict__ w1, const float* __restrict__ b1,
    const float* __restrict__ g1, const float* __restrict__ be1,
    const float* __restrict__ w2, const float* __restrict__ b2,
    const float* __restrict__ g2, const float* __restrict__ be2,
    _Float16* __restrict__ h2out)
{
  __shared__ float sw2[KH_ * KH_];
  __shared__ float sp[7 * KH_];
  for (int i = threadIdx.x; i < KH_ * KH_; i += 256) sw2[i] = w2[i];
  for (int i = threadIdx.x; i < KH_; i += 256) {
    sp[i]           = w1[i];
    sp[KH_ + i]     = b1[i];
    sp[2*KH_ + i]   = g1[i];
    sp[3*KH_ + i]   = be1[i];
    sp[4*KH_ + i]   = b2[i];
    sp[5*KH_ + i]   = g2[i];
    sp[6*KH_ + i]   = be2[i];
  }
  __syncthreads();
  int e = blockIdx.x * 256 + threadIdx.x;
  if (e >= E_) return;
  float t = node_t[nidx[e]] - edge_t[e];

  float h[KH_];
  float m = 0.f;
  #pragma unroll
  for (int j = 0; j < KH_; ++j) { h[j] = t * sp[j] + sp[KH_ + j]; m += h[j]; }
  m *= (1.f / KH_);
  float v = 0.f;
  #pragma unroll
  for (int j = 0; j < KH_; ++j) { float d = h[j] - m; v += d * d; }
  v *= (1.f / KH_);
  float rs = rsqrtf(v + 1e-5f);
  #pragma unroll
  for (int j = 0; j < KH_; ++j)
    h[j] = sinf(OMEGA_ * ((h[j] - m) * rs * sp[2*KH_ + j] + sp[3*KH_ + j]));

  float g[KH_];
  #pragma unroll
  for (int j = 0; j < KH_; ++j) {
    float a = sp[4*KH_ + j];
    #pragma unroll
    for (int k = 0; k < KH_; ++k) a += h[k] * sw2[k * KH_ + j];
    g[j] = a;
  }
  m = 0.f;
  #pragma unroll
  for (int j = 0; j < KH_; ++j) m += g[j];
  m *= (1.f / KH_);
  v = 0.f;
  #pragma unroll
  for (int j = 0; j < KH_; ++j) { float d = g[j] - m; v += d * d; }
  v *= (1.f / KH_);
  rs = rsqrtf(v + 1e-5f);

  _Float16* o = h2out + (size_t)e * H_;
  #pragma unroll
  for (int j = 0; j < KH_; ++j)
    o[j] = (_Float16)sinf(OMEGA_ * ((g[j] - m) * rs * sp[5*KH_ + j] + sp[6*KH_ + j]));
  o[KH_] = (_Float16)1.0f;                 // bias channel
  #pragma unroll
  for (int j = KH_ + 1; j < H_; ++j) o[j] = (_Float16)0.0f;
}

// Gather embedding rows per edge and convert to f16 (contiguous [E,64]).
__global__ void gather_half(const float* __restrict__ emb, const int* __restrict__ nidx,
                            _Float16* __restrict__ xg)
{
  int t = blockIdx.x * 256 + threadIdx.x;
  if (t >= E_ * H_) return;
  int e = t >> 6, c = t & 63;
  xg[t] = (_Float16)emb[nidx[e] * H_ + c];
}

// Pre-pack W3' (w3 rows + b3 as K-channel 50) into per-lane WMMA B-fragment order:
// bp[ch][ht][lane][i] with B 32x16 layout: lanes 0-15 hold kLocal 0-7,16-23, lanes 16-31 kLocal 8-15,24-31,
// column n = lane&15. Lane's 16 f16 are contiguous -> two global_load_b128 per fragment.
__global__ void pack_w3(const float* __restrict__ w3, const float* __restrict__ b3,
                        _Float16* __restrict__ bp)
{
  int idx = blockIdx.x * 256 + threadIdx.x;
  if (idx >= NCHUNK_ * 4 * 32 * 16) return;
  int i    = idx & 15;
  int lane = (idx >> 4) & 31;
  int ht   = (idx >> 9) & 3;
  int ch   = idx >> 11;
  int k    = ch * 32 + ((i >> 3) << 4) + ((lane >> 4) << 3) + (i & 7);   // global K index (j*64 + kk)
  int h    = ht * 16 + (lane & 15);              // output channel
  int j    = k >> 6;
  int kk   = k & 63;
  float val = (j < KH_) ? w3[j * (H_ * H_) + h * H_ + kk] : b3[h * H_ + kk];
  bp[idx] = (_Float16)val;
}

// Stage 2: per wave, 32 edges. C[32,64] = (h2 (x) x)[32,3264] @ W3'[3264,64] via
// v_wmma_f32_16x16x32_f16, A fragments built in-register, then atomic scatter-add.
__global__ __launch_bounds__(256) void edge_wmma_scatter(
    const _Float16* __restrict__ h2buf, const _Float16* __restrict__ xg,
    const _Float16* __restrict__ bp, const int* __restrict__ scat,
    float* __restrict__ outp)
{
  constexpr int XS = 72;                         // padded x row stride (bank-conflict free)
  __shared__ alignas(16) _Float16 sx[8][32 * XS];
  __shared__ alignas(16) _Float16 sh[8][64 * 32];  // h2 transposed: [j][row]
  __shared__ int sidx[8][32];

  const int lane = threadIdx.x & 31;
  const int w    = threadIdx.x >> 5;
  const int tile = blockIdx.x * 8 + w;
  const bool act = (tile < NT32_);
  const int e0   = tile * 32;

  if (act) {
    const uint4* gx = (const uint4*)(xg + (size_t)e0 * H_);
    const uint4* gh = (const uint4*)(h2buf + (size_t)e0 * H_);
    #pragma unroll
    for (int t2 = 0; t2 < 8; ++t2) {
      int idx = lane + 32 * t2;                  // uint4 (8xf16) index
      int row = idx >> 3;
      int c0  = (idx & 7) * 8;
      uint4 vx = gx[idx];
      *(uint4*)(&sx[w][row * XS + c0]) = vx;
      union { uint4 u; _Float16 f[8]; } cv;
      cv.u = gh[idx];
      #pragma unroll
      for (int q = 0; q < 8; ++q) sh[w][(c0 + q) * 32 + row] = cv.f[q];
    }
    int e = e0 + lane;
    sidx[w][lane] = (e < E_) ? scat[e] : 0;
  }
  __syncthreads();
  if (!act) return;

  const int m  = lane & 15;
  const int hi = lane >> 4;                      // A layout: lanes<16 K 0-7,16-23; >=16 K 8-15,24-31
  const _Float16* x0 = &sx[w][m * XS];
  const _Float16* x1 = &sx[w][(16 + m) * XS];

  v8f a00 = {}, a01 = {}, a02 = {}, a03 = {};
  v8f a10 = {}, a11 = {}, a12 = {}, a13 = {};

  for (int ch = 0; ch < NCHUNK_; ++ch) {
    const int j  = ch >> 1;
    const int k0 = (ch & 1) << 5;
    const _Float16 s0 = sh[w][j * 32 + m];
    const _Float16 s1 = sh[w][j * 32 + 16 + m];
    v8h p0 = *(const v8h*)(x0 + k0 + hi * 8);
    v8h p1 = *(const v8h*)(x0 + k0 + 16 + hi * 8);
    v8h q0 = *(const v8h*)(x1 + k0 + hi * 8);
    v8h q1 = *(const v8h*)(x1 + k0 + 16 + hi * 8);
    v16h fa, fb;
    #pragma unroll
    for (int q = 0; q < 8; ++q) {
      fa[q] = p0[q] * s0; fa[8 + q] = p1[q] * s0;
      fb[q] = q0[q] * s1; fb[8 + q] = q1[q] * s1;
    }
    const _Float16* bbase = bp + ((size_t)ch * 4 * 32) * 16;
    __builtin_prefetch(bp + ((size_t)(ch + 1) * 4 * 32) * 16, 0, 0);
    v16h b0 = *(const v16h*)(bbase + (0 * 32 + lane) * 16);
    v16h b1 = *(const v16h*)(bbase + (1 * 32 + lane) * 16);
    v16h b2 = *(const v16h*)(bbase + (2 * 32 + lane) * 16);
    v16h b3 = *(const v16h*)(bbase + (3 * 32 + lane) * 16);
    a00 = __builtin_amdgcn_wmma_f32_16x16x32_f16(false, fa, false, b0, (short)0, a00, false, false);
    a01 = __builtin_amdgcn_wmma_f32_16x16x32_f16(false, fa, false, b1, (short)0, a01, false, false);
    a02 = __builtin_amdgcn_wmma_f32_16x16x32_f16(false, fa, false, b2, (short)0, a02, false, false);
    a03 = __builtin_amdgcn_wmma_f32_16x16x32_f16(false, fa, false, b3, (short)0, a03, false, false);
    a10 = __builtin_amdgcn_wmma_f32_16x16x32_f16(false, fb, false, b0, (short)0, a10, false, false);
    a11 = __builtin_amdgcn_wmma_f32_16x16x32_f16(false, fb, false, b1, (short)0, a11, false, false);
    a12 = __builtin_amdgcn_wmma_f32_16x16x32_f16(false, fb, false, b2, (short)0, a12, false, false);
    a13 = __builtin_amdgcn_wmma_f32_16x16x32_f16(false, fb, false, b3, (short)0, a13, false, false);
  }

  // C layout: VGPR r, lanes<16 -> M=r, lanes>=16 -> M=r+8; N = lane&15.
  const int n = lane & 15;
  #pragma unroll
  for (int r = 0; r < 8; ++r) {
    int ma = r + hi * 8;
    int ea = e0 + ma;
    if (ea < E_) {
      float* dst = outp + (size_t)sidx[w][ma] * H_ + n;
      atomicAdd(dst + 0,  a00[r]);
      atomicAdd(dst + 16, a01[r]);
      atomicAdd(dst + 32, a02[r]);
      atomicAdd(dst + 48, a03[r]);
    }
    int eb = e0 + 16 + ma;
    if (eb < E_) {
      float* dst = outp + (size_t)sidx[w][16 + ma] * H_ + n;
      atomicAdd(dst + 0,  a10[r]);
      atomicAdd(dst + 16, a11[r]);
      atomicAdd(dst + 32, a12[r]);
      atomicAdd(dst + 48, a13[r]);
    }
  }
}

extern "C" void kernel_launch(void* const* d_in, const int* in_sizes, int n_in,
                              void* d_out, int out_size, void* d_ws, size_t ws_size,
                              hipStream_t stream)
{
  (void)in_sizes; (void)n_in; (void)ws_size;
  const float* u_emb  = (const float*)d_in[0];
  const float* i_emb  = (const float*)d_in[1];
  const int*   uix    = (const int*)d_in[2];
  const int*   iix    = (const int*)d_in[3];
  const float* edge_t = (const float*)d_in[4];
  const float* u_t    = (const float*)d_in[5];
  const float* i_t    = (const float*)d_in[6];
  const float* up[10]; const float* ip[10];
  for (int q = 0; q < 10; ++q) { up[q] = (const float*)d_in[7 + q]; ip[q] = (const float*)d_in[17 + q]; }

  char* ws = (char*)d_ws;
  size_t off = 0;
  auto take = [&](size_t bytes) { void* p = ws + off; off += (bytes + 255) & ~(size_t)255; return p; };
  _Float16* h2_i = (_Float16*)take((size_t)EPAD_ * H_ * 2);
  _Float16* h2_u = (_Float16*)take((size_t)EPAD_ * H_ * 2);
  _Float16* xg_i = (_Float16*)take((size_t)EPAD_ * H_ * 2);
  _Float16* xg_u = (_Float16*)take((size_t)EPAD_ * H_ * 2);
  _Float16* bp_i = (_Float16*)take((size_t)NCHUNK_ * 2048 * 2);
  _Float16* bp_u = (_Float16*)take((size_t)NCHUNK_ * 2048 * 2);

  float* out = (float*)d_out;
  zero_out<<<(out_size + 255) / 256, 256, 0, stream>>>(out, out_size);

  knet_h2<<<(E_ + 255) / 256, 256, 0, stream>>>(i_t, edge_t, iix,
      ip[0], ip[1], ip[2], ip[3], ip[4], ip[5], ip[6], ip[7], h2_i);
  knet_h2<<<(E_ + 255) / 256, 256, 0, stream>>>(u_t, edge_t, uix,
      up[0], up[1], up[2], up[3], up[4], up[5], up[6], up[7], h2_u);
  gather_half<<<(E_ * H_ + 255) / 256, 256, 0, stream>>>(i_emb, iix, xg_i);
  gather_half<<<(E_ * H_ + 255) / 256, 256, 0, stream>>>(u_emb, uix, xg_u);
  const int pk = NCHUNK_ * 2048;
  pack_w3<<<(pk + 255) / 256, 256, 0, stream>>>(ip[8], ip[9], bp_i);
  pack_w3<<<(pk + 255) / 256, 256, 0, stream>>>(up[8], up[9], bp_u);

  const int blocks2 = (NT32_ + 7) / 8;
  // item kernels x item embeddings -> scatter by user into hLu (out[0 .. NU*64))
  edge_wmma_scatter<<<blocks2, 256, 0, stream>>>(h2_i, xg_i, bp_i, uix, out);
  // user kernels x user embeddings -> scatter by item into hLi (out[NU*64 ..))
  edge_wmma_scatter<<<blocks2, 256, 0, stream>>>(h2_u, xg_u, bp_u, iix, out + (size_t)NU_ * H_);
}
